// TopKGate_51977694216448
// MI455X (gfx1250) — compile-verified, matrix operand order checked by reading, not run on hardware
//
#include <hip/hip_runtime.h>
#include <hip/hip_fp16.h>

typedef __attribute__((ext_vector_type(16))) _Float16     v16h;
typedef __attribute__((ext_vector_type(8)))  float        v8f;
typedef __attribute__((ext_vector_type(4)))  float        f4;
typedef __attribute__((ext_vector_type(4)))  unsigned int u4;

#define S_ROWS 16384
#define C_DIM  2048
#define E_DIM  64

union FragH {
    v16h v;
    u4   q[2];
    _Float16 h[16];
};

// Pack two fp32 -> two fp16 via v_cvt_pk_rtz_f16_f32 (1 VALU per 2 elems).
__device__ __forceinline__ void pack2(FragH& f, int idx, float lo, float hi) {
    auto p = __builtin_amdgcn_cvt_pkrtz(lo, hi);
    __builtin_memcpy(&f.h[idx], &p, 4);
}

__device__ __forceinline__ FragH cvt_frag(f4 y0, f4 y1, f4 y2, f4 y3) {
    FragH f;
    pack2(f, 0,  y0[0], y0[1]); pack2(f, 2,  y0[2], y0[3]);
    pack2(f, 4,  y1[0], y1[1]); pack2(f, 6,  y1[2], y1[3]);
    pack2(f, 8,  y2[0], y2[1]); pack2(f, 10, y2[2], y2[3]);
    pack2(f, 12, y3[0], y3[1]); pack2(f, 14, y3[2], y3[3]);
    return f;
}

__global__ void zero_ws_kernel(float* __restrict__ ws) {
    int t = threadIdx.x;
    if (t < 128) ws[t] = 0.0f;
}

// One-time W fp32 -> fp16 (64 x 2048: 512KB -> 256KB, L2-resident).
__global__ void convert_W_kernel(const float* __restrict__ W,
                                 _Float16* __restrict__ Wh) {
    int i = blockIdx.x * blockDim.x + threadIdx.x;
    if (i < E_DIM * C_DIM) Wh[i] = (_Float16)W[i];
}

// Block: 128 threads = 4 waves. Each wave computes a 16(M) x 64(E) logits tile
// via v_wmma_f32_16x16x32_f16 over C=2048 (64 K-steps), 2-stage software
// pipeline (loads 2 ahead, conversion 1 ahead), then softmax + top-2 + aux.
__global__ __launch_bounds__(128) void gate_gemm_kernel(
    const float* __restrict__ x, const _Float16* __restrict__ Wh,
    const float* __restrict__ bias, float* __restrict__ out,
    float* __restrict__ ws) {

    __shared__ float lds_logits[64 * 64];   // 64 rows x 64 experts
    __shared__ float lds_imp[E_DIM];
    __shared__ float lds_cnt[E_DIM];

    const int tid  = threadIdx.x;
    const int wave = tid >> 5;
    const int lane = tid & 31;

    if (tid < E_DIM) { lds_imp[tid] = 0.0f; lds_cnt[tid] = 0.0f; }

    const int row_base = blockIdx.x * 64 + wave * 16;
    const int m  = lane & 15;          // row (A) / col (B) within the 16-tile
    const int kb = (lane >> 4) * 8;    // per-lane K sub-offset per ISA layout

    const float* xrow = x + (size_t)(row_base + m) * C_DIM + kb;
    const _Float16* w0 = Wh + (size_t)( 0 + m) * C_DIM + kb;
    const _Float16* w1 = Wh + (size_t)(16 + m) * C_DIM + kb;
    const _Float16* w2 = Wh + (size_t)(32 + m) * C_DIM + kb;
    const _Float16* w3 = Wh + (size_t)(48 + m) * C_DIM + kb;

    v8f acc0 = {}, acc1 = {}, acc2 = {}, acc3 = {};

    // ---- 2-stage software pipeline state ----
    FragH aCur;                         // converted A for step k
    f4 f0, f1, f2, f3;                  // A floats for step k+32 (in flight)
    u4 b00, b01, b10, b11, b20, b21, b30, b31;  // B raw for step k
    u4 n00, n01, n10, n11, n20, n21, n30, n31;  // B raw for step k+32

    {   // prologue: k = 0 operands + k = 32 loads
        f4 t0 = *(const f4*)(xrow);      f4 t1 = *(const f4*)(xrow + 4);
        f4 t2 = *(const f4*)(xrow + 16); f4 t3 = *(const f4*)(xrow + 20);
        b00 = *(const u4*)(w0);  b01 = *(const u4*)(w0 + 16);
        b10 = *(const u4*)(w1);  b11 = *(const u4*)(w1 + 16);
        b20 = *(const u4*)(w2);  b21 = *(const u4*)(w2 + 16);
        b30 = *(const u4*)(w3);  b31 = *(const u4*)(w3 + 16);
        aCur = cvt_frag(t0, t1, t2, t3);

        f0 = *(const f4*)(xrow + 32);      f1 = *(const f4*)(xrow + 36);
        f2 = *(const f4*)(xrow + 48); f3 = *(const f4*)(xrow + 52);
        n00 = *(const u4*)(w0 + 32);  n01 = *(const u4*)(w0 + 48);
        n10 = *(const u4*)(w1 + 32);  n11 = *(const u4*)(w1 + 48);
        n20 = *(const u4*)(w2 + 32);  n21 = *(const u4*)(w2 + 48);
        n30 = *(const u4*)(w3 + 32);  n31 = *(const u4*)(w3 + 48);
    }

    for (int k0 = 0; k0 < C_DIM; k0 += 32) {
        // Branchless clamped offsets (tail iterations reload in-bounds data).
        const int k2 = (k0 + 64 < C_DIM) ? (k0 + 64) : k0;   // loads: 2 ahead
        const int kp = (k0 + 96 < C_DIM) ? (k0 + 96) : k0;   // prefetch: 3 ahead
        __builtin_prefetch(xrow + kp, 0, 3);                  // global_prefetch_b8

        // Stage 1: issue loads two steps ahead.
        f4 g0 = *(const f4*)(xrow + k2);
        f4 g1 = *(const f4*)(xrow + k2 + 4);
        f4 g2 = *(const f4*)(xrow + k2 + 16);
        f4 g3 = *(const f4*)(xrow + k2 + 20);
        u4 m00 = *(const u4*)(w0 + k2), m01 = *(const u4*)(w0 + k2 + 16);
        u4 m10 = *(const u4*)(w1 + k2), m11 = *(const u4*)(w1 + k2 + 16);
        u4 m20 = *(const u4*)(w2 + k2), m21 = *(const u4*)(w2 + k2 + 16);
        u4 m30 = *(const u4*)(w3 + k2), m31 = *(const u4*)(w3 + k2 + 16);

        // Stage 2: convert next step's A (arrived last iteration) into a
        // fragment register set distinct from the one the WMMAs read below.
        FragH aNext = cvt_frag(f0, f1, f2, f3);

        // Stage 3: matrix ops on current operands.
        FragH B0; B0.q[0] = b00; B0.q[1] = b01;
        FragH B1; B1.q[0] = b10; B1.q[1] = b11;
        FragH B2; B2.q[0] = b20; B2.q[1] = b21;
        FragH B3; B3.q[0] = b30; B3.q[1] = b31;

        acc0 = __builtin_amdgcn_wmma_f32_16x16x32_f16(false, aCur.v, false, B0.v,
                                                      (short)0, acc0, false, false);
        acc1 = __builtin_amdgcn_wmma_f32_16x16x32_f16(false, aCur.v, false, B1.v,
                                                      (short)0, acc1, false, false);
        acc2 = __builtin_amdgcn_wmma_f32_16x16x32_f16(false, aCur.v, false, B2.v,
                                                      (short)0, acc2, false, false);
        acc3 = __builtin_amdgcn_wmma_f32_16x16x32_f16(false, aCur.v, false, B3.v,
                                                      (short)0, acc3, false, false);

        // Shift pipeline (renamed away by unrolling).
        aCur = aNext;
        f0 = g0; f1 = g1; f2 = g2; f3 = g3;
        b00 = n00; b01 = n01; b10 = n10; b11 = n11;
        b20 = n20; b21 = n21; b30 = n30; b31 = n31;
        n00 = m00; n01 = m01; n10 = m10; n11 = m11;
        n20 = m20; n21 = m21; n30 = m30; n31 = m31;
    }

    // C/D layout: lane 0-15 -> N=lane, M=r; lane 16-31 -> N=lane-16, M=r+8
    {
        const int half = (lane >> 4) * 8;
        float bb0 = bias[ 0 + m], bb1 = bias[16 + m];
        float bb2 = bias[32 + m], bb3 = bias[48 + m];
#pragma unroll
        for (int r = 0; r < 8; ++r) {
            int row = wave * 16 + half + r;
            lds_logits[row * 64 +  0 + m] = acc0[r] + bb0;
            lds_logits[row * 64 + 16 + m] = acc1[r] + bb1;
            lds_logits[row * 64 + 32 + m] = acc2[r] + bb2;
            lds_logits[row * 64 + 48 + m] = acc3[r] + bb3;
        }
    }
    __syncthreads();

    // Softmax + top-2 per row; threads 0..63 each own one of the 64 rows.
    if (tid < 64) {
        const float* lrow = lds_logits + tid * 64;
        float m1 = -1e30f; int i1 = 0;
#pragma unroll 8
        for (int e = 0; e < E_DIM; ++e) {
            float l = lrow[e];
            if (l > m1) { m1 = l; i1 = e; }
        }
        float m2 = -1e30f; int i2 = (i1 == 0) ? 1 : 0;
#pragma unroll 8
        for (int e = 0; e < E_DIM; ++e) {
            float l = lrow[e];
            if (e != i1 && l > m2) { m2 = l; i2 = e; }
        }
        float sum = 0.0f;
#pragma unroll 8
        for (int e = 0; e < E_DIM; ++e) sum += __expf(lrow[e] - m1);
        float inv = 1.0f / sum;

        int grow = blockIdx.x * 64 + tid;
        out[grow * 2 + 0] = (float)i1;                       // topk_idx
        out[grow * 2 + 1] = (float)i2;
        out[2 * S_ROWS + grow * 2 + 0] = inv;                // topk_vals: exp(0)/sum
        out[2 * S_ROWS + grow * 2 + 1] = __expf(m2 - m1) * inv;

        // importance accumulation (ds_add_f32) + argmax histogram
#pragma unroll 8
        for (int e = 0; e < E_DIM; ++e) {
            atomicAdd(&lds_imp[e], __expf(lrow[e] - m1) * inv);
        }
        atomicAdd(&lds_cnt[i1], 1.0f);
    }
    __syncthreads();

    if (tid < E_DIM) {
        atomicAdd(&ws[tid],          lds_imp[tid]);   // global_atomic_add_f32
        atomicAdd(&ws[E_DIM + tid],  lds_cnt[tid]);
    }
}

__global__ void finalize_kernel(const float* __restrict__ ws,
                                float* __restrict__ out) {
    __shared__ float red[E_DIM];
    int t = threadIdx.x;
    float invS = 1.0f / (float)S_ROWS;
    red[t] = (ws[t] * invS) * (ws[E_DIM + t] * invS);
    __syncthreads();
    if (t == 0) {
        float s = 0.0f;
#pragma unroll
        for (int e = 0; e < E_DIM; ++e) s += red[e];
        out[4 * S_ROWS] = (float)E_DIM * s;   // aux_loss after idx + vals
    }
}

extern "C" void kernel_launch(void* const* d_in, const int* in_sizes, int n_in,
                              void* d_out, int out_size, void* d_ws, size_t ws_size,
                              hipStream_t stream) {
    (void)in_sizes; (void)n_in; (void)out_size; (void)ws_size;
    const float* x    = (const float*)d_in[0];
    const float* W    = (const float*)d_in[1];
    const float* bias = (const float*)d_in[2];
    float* out = (float*)d_out;
    float* ws  = (float*)d_ws;

    // ws layout: [0..127] float accumulators (imp, cnt), then fp16 W (256KB).
    _Float16* Wh = (_Float16*)(ws + 128);

    zero_ws_kernel<<<1, 128, 0, stream>>>(ws);
    convert_W_kernel<<<(E_DIM * C_DIM + 255) / 256, 256, 0, stream>>>(W, Wh);
    gate_gemm_kernel<<<S_ROWS / 64, 128, 0, stream>>>(x, Wh, bias, out, ws);
    finalize_kernel<<<1, E_DIM, 0, stream>>>(ws, out);
}
